// DependencyNeuralModel_21766894256588
// MI455X (gfx1250) — matrix operand order repaired and checked by reading, not compiled, not run on hardware
//
#include <hip/hip_runtime.h>
#include <hip/hip_bf16.h>

// ---------------------------------------------------------------------------
// Dimensions
// ---------------------------------------------------------------------------
#define S_LEN   512
#define N_ARC   (S_LEN * S_LEN)      // 262144
#define WE_DIM  300
#define TE_DIM  64
#define IN_DIM  364                  // WE + TE
#define IN_PAD  384                  // padded to multiple of 32
#define H_DIM   600
#define G4_DIM  2400                 // 4*H
#define K2_PAD  1216                 // 2*H padded to multiple of 32
#define N_PAD   608                  // H padded to multiple of 32
#define DE_DIM  128
#define NBINS   17
#define NB2     34                   // 2*NBINS
#define MD_PAD  48                   // NB2 padded to multiple of 16

typedef __attribute__((ext_vector_type(16))) _Float16     v16h;
typedef __attribute__((ext_vector_type(8)))  float        v8f;
typedef __attribute__((ext_vector_type(4)))  unsigned int u32x4;
typedef __attribute__((ext_vector_type(8)))  int          i32x8;
typedef __attribute__((ext_vector_type(4)))  int          i32x4;

// ---------------------------------------------------------------------------
// f32 -> f16 convert with zero padding (fully rewrites dst incl. pad region)
// ---------------------------------------------------------------------------
__global__ void k_cvt_pad(const float* __restrict__ src, _Float16* __restrict__ dst,
                          int srcRows, int srcCols, int dstRows, int dstCols) {
    int idx = blockIdx.x * blockDim.x + threadIdx.x;
    int total = dstRows * dstCols;
    if (idx >= total) return;
    int r = idx / dstCols, c = idx - r * dstCols;
    float v = (r < srcRows && c < srcCols) ? src[(size_t)r * srcCols + c] : 0.0f;
    dst[idx] = (_Float16)v;
}

// ---------------------------------------------------------------------------
// Embedding gather: x16[512,384] = concat(word_emb[words], tag_emb[tags]), pad 0
// ---------------------------------------------------------------------------
__global__ void k_embed(const int* __restrict__ words, const int* __restrict__ tags,
                        const float* __restrict__ wemb, const float* __restrict__ temb,
                        _Float16* __restrict__ x16) {
    int idx = blockIdx.x * blockDim.x + threadIdx.x;
    if (idx >= S_LEN * IN_PAD) return;
    int r = idx / IN_PAD, c = idx - r * IN_PAD;
    float v = 0.0f;
    if (c < WE_DIM)          v = wemb[(size_t)words[r] * WE_DIM + c];
    else if (c < IN_DIM)     v = temb[(size_t)tags[r] * TE_DIM + (c - WE_DIM)];
    x16[idx] = (_Float16)v;
}

// Zero the K padding columns (1200..1215) of states16
__global__ void k_states_pad(_Float16* __restrict__ s16) {
    int idx = blockIdx.x * blockDim.x + threadIdx.x;
    if (idx >= S_LEN * 16) return;
    int r = idx >> 4, c = idx & 15;
    s16[(size_t)r * K2_PAD + 2 * H_DIM + c] = (_Float16)0.0f;
}

// ---------------------------------------------------------------------------
// WMMA GEMM:  out[M, Np] (f32) = A[M, Kp] (f16) x W[Np, Kp]^T (f16)  (+ bias)
// One wave per 16x32 output strip (2 N-tiles) -> A-fragment reused by two
// back-to-back v_wmma issues per K step.
// M % 16 == 0, Np % 32 == 0, Kp % 32 == 0.
// ---------------------------------------------------------------------------
__global__ void k_wmma_gemm(const _Float16* __restrict__ A, const _Float16* __restrict__ W,
                            float* __restrict__ out, int M, int Np, int Kp,
                            const float* __restrict__ bias0,
                            const float* __restrict__ bias1, int nBias) {
    int wave = (blockIdx.x * blockDim.x + threadIdx.x) >> 5;
    int lane = threadIdx.x & 31;
    int stripsN = Np >> 5;                     // 32-column strips
    int tiles   = (M >> 4) * stripsN;
    if (wave >= tiles) return;                 // whole-wave uniform exit
    int tm = wave / stripsN, tn = wave - tm * stripsN;
    int lHalf = lane >> 4, lMod = lane & 15;

    const unsigned int* Arow  = (const unsigned int*)(A + (size_t)(tm * 16 + lMod) * Kp);
    const unsigned int* Wrow0 = (const unsigned int*)(W + (size_t)(tn * 32 + lMod) * Kp);
    const unsigned int* Wrow1 = (const unsigned int*)(W + (size_t)(tn * 32 + 16 + lMod) * Kp);

    v8f c0 = {}, c1 = {};
    for (int kb = 0; kb < Kp; kb += 32) {
        union { unsigned int u[8]; v16h v; } a, b0, b1;
#pragma unroll
        for (int r = 0; r < 8; ++r) {
            // A operand, contiguous k order used here (same as B): VGPR r holds K pair at 2r + 16*half
            int ka = 2 * r + 16 * lHalf;
            // B operand, contiguous k order used here (same as A): VGPR r holds K pair at 2r + 16*half
            int kw = 2 * r + 16 * lHalf;
            a.u[r]  = Arow [(kb + ka) >> 1];
            b0.u[r] = Wrow0[(kb + kw) >> 1];
            b1.u[r] = Wrow1[(kb + kw) >> 1];
        }
        c0 = __builtin_amdgcn_wmma_f32_16x16x32_f16(false, a.v, false, b0.v,
                                                    (short)0, c0, false, false);
        c1 = __builtin_amdgcn_wmma_f32_16x16x32_f16(false, a.v, false, b1.v,
                                                    (short)0, c1, false, false);
    }

    int n0 = tn * 32 + lMod;
    int n1 = n0 + 16;
    float bv0 = 0.0f, bv1 = 0.0f;
    if (n0 < nBias) { if (bias0) bv0 += bias0[n0]; if (bias1) bv0 += bias1[n0]; }
    if (n1 < nBias) { if (bias0) bv1 += bias0[n1]; if (bias1) bv1 += bias1[n1]; }
#pragma unroll
    for (int r = 0; r < 8; ++r) {
        int m = tm * 16 + r + 8 * lHalf;       // C/D layout: VGPR r -> M=r / M=8+r
        out[(size_t)m * Np + n0] = c0[r] + bv0;
        out[(size_t)m * Np + n1] = c1[r] + bv1;
    }
}

// ---------------------------------------------------------------------------
// Bidirectional LSTM scan. grid.x = 2 (dir), block = 800 threads.
// gx[t] precomputed (x@W_ih^T + b_ih + b_hh). Gate order: i, f, g, o.
// h broadcast from LDS (b128 reads); W_hh rows streamed from L2 as float4.
// Next timestep's gx row is prefetched (global_prefetch_b8).
// ---------------------------------------------------------------------------
__global__ __launch_bounds__(800) void k_lstm(const float* __restrict__ gx_f,
                                              const float* __restrict__ gx_b,
                                              const float* __restrict__ whh_f,
                                              const float* __restrict__ whh_b,
                                              _Float16* __restrict__ states16) {
    __shared__ __align__(16) float h_lds[H_DIM];
    __shared__ float g_lds[G4_DIM];
    const int dir = blockIdx.x;
    const float* gx  = dir ? gx_b  : gx_f;
    const float* whh = dir ? whh_b : whh_f;
    const int tid = threadIdx.x;

    float c_st = 0.0f;
    if (tid < H_DIM) h_lds[tid] = 0.0f;
    __syncthreads();

    for (int step = 0; step < S_LEN; ++step) {
        const int t = dir ? (S_LEN - 1 - step) : step;
        const float* gxt = gx + (size_t)t * G4_DIM;
        if (step + 1 < S_LEN) {
            const int tn = dir ? (S_LEN - 2 - step) : (step + 1);
            const float* gxn = gx + (size_t)tn * G4_DIM;
#pragma unroll
            for (int rr = 0; rr < 3; ++rr)
                __builtin_prefetch(gxn + tid + rr * 800, 0, 0);
        }
        // Phase A: 800 threads x 3 rows => 2400 gate pre-activations
#pragma unroll
        for (int rr = 0; rr < 3; ++rr) {
            const int row = tid + rr * 800;
            const float4* wrow4 = (const float4*)(whh + (size_t)row * H_DIM);
            const float4* h4 = (const float4*)h_lds;
            float acc = gxt[row];
#pragma unroll 2
            for (int k = 0; k < H_DIM / 4; ++k) {
                float4 w = wrow4[k], h = h4[k];
                acc = fmaf(w.x, h.x, acc);
                acc = fmaf(w.y, h.y, acc);
                acc = fmaf(w.z, h.z, acc);
                acc = fmaf(w.w, h.w, acc);
            }
            g_lds[row] = acc;
        }
        __syncthreads();
        // Phase B: state update (threads 0..599 own one hidden unit each)
        if (tid < H_DIM) {
            float ig = 1.0f / (1.0f + __expf(-g_lds[tid]));
            float fg = 1.0f / (1.0f + __expf(-g_lds[tid + H_DIM]));
            float gg = tanhf(g_lds[tid + 2 * H_DIM]);
            float og = 1.0f / (1.0f + __expf(-g_lds[tid + 3 * H_DIM]));
            c_st = fg * c_st + ig * gg;
            float h = og * tanhf(c_st);
            h_lds[tid] = h;
            states16[(size_t)t * K2_PAD + dir * H_DIM + tid] = (_Float16)h;
        }
        __syncthreads();
    }
}

// ---------------------------------------------------------------------------
// Arc scoring: one wave per arc. arc_w staged into LDS once per block via the
// Tensor Data Mover (1-D D# descriptor), wave 0 issues, s_wait_tensorcnt 0.
// score[a] = sum_j arc_w[j] * tanh(heads[h][j] + mods[m][j] + dtab[d][j])
// ---------------------------------------------------------------------------
__global__ void k_arc(const int* __restrict__ arcH, const int* __restrict__ arcM,
                      const float* __restrict__ Hd, const float* __restrict__ Md,
                      const float* __restrict__ Dt, const float* __restrict__ arc_w,
                      float* __restrict__ scores) {
    __shared__ __align__(16) float aw[N_PAD];

#if __has_builtin(__builtin_amdgcn_tensor_load_to_lds)
    if (threadIdx.x < 32) {                    // uniform branch: only wave 0 issues TDM
        unsigned long long ga = (unsigned long long)arc_w;
        unsigned int ldsOff = (unsigned int)(unsigned long long)(void*)aw;
        // D# group 0: count=1 | lds_addr | global_addr[56:0] | type=2
        u32x4 g0 = { 1u, ldsOff, (unsigned int)ga,
                     (unsigned int)((ga >> 32) & 0x01FFFFFFu) | 0x80000000u };
        // D# group 1: data_size=4B; tensor_dim0=600, tensor_dim1=1;
        //             tile_dim0=600, tile_dim1=1; strides=600
        i32x8 g1 = { (int)0x00020000u,              // workgroup_mask=0, data_size=2(4B)
                     (int)((unsigned)H_DIM << 16),  // tensor_dim0 lo16
                     (int)(1u << 16),               // tensor_dim0 hi=0, tensor_dim1 lo16=1
                     (int)((unsigned)H_DIM << 16),  // tensor_dim1 hi=0, tile_dim0=600
                     1,                             // tile_dim1=1, tile_dim2=0
                     H_DIM,                         // tensor_dim0_stride lo32
                     (int)((unsigned)H_DIM << 16),  // stride0 hi=0, stride1 lo16
                     0 };                           // stride1 hi
        i32x4 z4 = {};
#if defined(__clang_major__) && (__clang_major__ >= 23)
        i32x8 z8 = {};
        __builtin_amdgcn_tensor_load_to_lds(g0, g1, z4, z4, z8, 0);
#else
        __builtin_amdgcn_tensor_load_to_lds(g0, g1, z4, z4, 0);
#endif
        __builtin_amdgcn_s_wait_tensorcnt(0);
    }
#else
    for (int j = threadIdx.x; j < H_DIM; j += blockDim.x) aw[j] = arc_w[j];
#endif
    __syncthreads();

    int wave = (blockIdx.x * blockDim.x + threadIdx.x) >> 5;
    int lane = threadIdx.x & 31;
    if (wave >= N_ARC) return;
    int h = arcH[wave], m = arcM[wave];
    int diff = m - h;
    int absd = diff < 0 ? -diff : diff;
    // bins = [0..9, 10, 15, 20, 25, 30, 35, 40]; last bin index <= absd
    int bin = (absd < 10) ? absd : ((absd < 40) ? (10 + (absd - 10) / 5) : 16);
    int d = (diff > 0) ? bin : (bin + NBINS);
    const float* hp = Hd + (size_t)h * N_PAD;
    const float* mp = Md + (size_t)m * N_PAD;
    const float* dp = Dt + (size_t)d * N_PAD;
    float s = 0.0f;
    for (int j = lane; j < H_DIM; j += 32)
        s += aw[j] * tanhf(hp[j] + mp[j] + dp[j]);
#pragma unroll
    for (int off = 16; off > 0; off >>= 1) s += __shfl_down(s, off, 32);
    if (lane == 0) scores[wave] = s;
}

// ---------------------------------------------------------------------------
// Launcher
// ---------------------------------------------------------------------------
extern "C" void kernel_launch(void* const* d_in, const int* in_sizes, int n_in,
                              void* d_out, int out_size, void* d_ws, size_t ws_size,
                              hipStream_t stream) {
    const int*   words    = (const int*)  d_in[0];
    const int*   tags     = (const int*)  d_in[1];
    const int*   arcH     = (const int*)  d_in[2];
    const int*   arcM     = (const int*)  d_in[3];
    const float* word_emb = (const float*)d_in[4];
    const float* tag_emb  = (const float*)d_in[5];
    const float* dist_emb = (const float*)d_in[6];
    const float* w_ih_f   = (const float*)d_in[7];
    const float* w_hh_f   = (const float*)d_in[8];
    const float* b_ih_f   = (const float*)d_in[9];
    const float* b_hh_f   = (const float*)d_in[10];
    const float* w_ih_b   = (const float*)d_in[11];
    const float* w_hh_b   = (const float*)d_in[12];
    const float* b_ih_b   = (const float*)d_in[13];
    const float* b_hh_b   = (const float*)d_in[14];
    const float* head_W   = (const float*)d_in[15];
    const float* mod_W    = (const float*)d_in[16];
    const float* dist_W   = (const float*)d_in[17];
    const float* dist_b   = (const float*)d_in[18];
    const float* arc_w    = (const float*)d_in[19];
    float* scores = (float*)d_out;

    // ---- workspace layout (all sizes are multiples of 256 bytes) ----
    char* ws = (char*)d_ws;
    size_t off = 0;
    auto carve = [&](size_t bytes) { void* p = ws + off; off += bytes; return p; };
    _Float16* x16     = (_Float16*)carve((size_t)S_LEN * IN_PAD * 2);      // 384 KB
    _Float16* wihf16  = (_Float16*)carve((size_t)G4_DIM * IN_PAD * 2);     // 1.8 MB
    _Float16* wihb16  = (_Float16*)carve((size_t)G4_DIM * IN_PAD * 2);
    _Float16* hW16    = (_Float16*)carve((size_t)N_PAD * K2_PAD * 2);      // 1.4 MB
    _Float16* mW16    = (_Float16*)carve((size_t)N_PAD * K2_PAD * 2);
    _Float16* dE16    = (_Float16*)carve((size_t)MD_PAD * DE_DIM * 2);     // 12 KB
    _Float16* dW16    = (_Float16*)carve((size_t)N_PAD * DE_DIM * 2);      // 152 KB
    float*    gx_f    = (float*)   carve((size_t)S_LEN * G4_DIM * 4);      // 4.9 MB
    float*    gx_b    = (float*)   carve((size_t)S_LEN * G4_DIM * 4);
    _Float16* st16    = (_Float16*)carve((size_t)S_LEN * K2_PAD * 2);      // 1.2 MB
    float*    headsT  = (float*)   carve((size_t)S_LEN * N_PAD * 4);       // 1.2 MB
    float*    modsT   = (float*)   carve((size_t)S_LEN * N_PAD * 4);
    float*    dtab    = (float*)   carve((size_t)MD_PAD * N_PAD * 4);      // 114 KB
    (void)ws_size; (void)in_sizes; (void)n_in; (void)out_size;

    const int TB = 256;
    auto blocks = [](long long n, int tb) { return (unsigned)((n + tb - 1) / tb); };

    // 1) Weight conversions + embedding gather (independent)
    k_cvt_pad<<<blocks((long long)G4_DIM * IN_PAD, TB), TB, 0, stream>>>(
        w_ih_f, wihf16, G4_DIM, IN_DIM, G4_DIM, IN_PAD);
    k_cvt_pad<<<blocks((long long)G4_DIM * IN_PAD, TB), TB, 0, stream>>>(
        w_ih_b, wihb16, G4_DIM, IN_DIM, G4_DIM, IN_PAD);
    k_cvt_pad<<<blocks((long long)N_PAD * K2_PAD, TB), TB, 0, stream>>>(
        head_W, hW16, H_DIM, 2 * H_DIM, N_PAD, K2_PAD);
    k_cvt_pad<<<blocks((long long)N_PAD * K2_PAD, TB), TB, 0, stream>>>(
        mod_W, mW16, H_DIM, 2 * H_DIM, N_PAD, K2_PAD);
    k_cvt_pad<<<blocks((long long)MD_PAD * DE_DIM, TB), TB, 0, stream>>>(
        dist_emb, dE16, NB2, DE_DIM, MD_PAD, DE_DIM);
    k_cvt_pad<<<blocks((long long)N_PAD * DE_DIM, TB), TB, 0, stream>>>(
        dist_W, dW16, H_DIM, DE_DIM, N_PAD, DE_DIM);
    k_embed<<<blocks((long long)S_LEN * IN_PAD, TB), TB, 0, stream>>>(
        words, tags, word_emb, tag_emb, x16);
    k_states_pad<<<blocks((long long)S_LEN * 16, TB), TB, 0, stream>>>(st16);

    // 2) gx GEMMs: [512,384] x [2400,384]^T -> [512,2400]  (bias = b_ih + b_hh)
    {
        int tiles = (S_LEN / 16) * (G4_DIM / 32);              // 2400 waves
        unsigned nb = blocks((long long)tiles * 32, TB);
        k_wmma_gemm<<<nb, TB, 0, stream>>>(x16, wihf16, gx_f, S_LEN, G4_DIM, IN_PAD,
                                           b_ih_f, b_hh_f, G4_DIM);
        k_wmma_gemm<<<nb, TB, 0, stream>>>(x16, wihb16, gx_b, S_LEN, G4_DIM, IN_PAD,
                                           b_ih_b, b_hh_b, G4_DIM);
    }

    // 3) Distance table GEMM: [48,128] x [608,128]^T -> [48,608]  (+ dist_b)
    {
        int tiles = (MD_PAD / 16) * (N_PAD / 32);              // 57 waves
        k_wmma_gemm<<<blocks((long long)tiles * 32, TB), TB, 0, stream>>>(
            dE16, dW16, dtab, MD_PAD, N_PAD, DE_DIM, dist_b, nullptr, H_DIM);
    }

    // 4) Bidirectional LSTM scan (2 blocks, one per direction)
    k_lstm<<<2, 800, 0, stream>>>(gx_f, gx_b, w_hh_f, w_hh_b, st16);

    // 5) Projections: [512,1216] x [608,1216]^T -> [512,608]
    {
        int tiles = (S_LEN / 16) * (N_PAD / 32);               // 608 waves
        unsigned nb = blocks((long long)tiles * 32, TB);
        k_wmma_gemm<<<nb, TB, 0, stream>>>(st16, hW16, headsT, S_LEN, N_PAD, K2_PAD,
                                           nullptr, nullptr, 0);
        k_wmma_gemm<<<nb, TB, 0, stream>>>(st16, mW16, modsT, S_LEN, N_PAD, K2_PAD,
                                           nullptr, nullptr, 0);
    }

    // 6) Arc scoring: one wave per arc (262144 waves)
    k_arc<<<blocks((long long)N_ARC * 32, TB), TB, 0, stream>>>(
        arcH, arcM, headsT, modsT, dtab, arc_w, scores);
}
